// AdaptiveUserAwareAttention_37890201485313
// MI455X (gfx1250) — matrix operand order repaired and checked by reading, not compiled, not run on hardware
//
#include <hip/hip_runtime.h>
#include <math.h>

#define B_ 4
#define S_ 1024
#define D_ 1024
#define H_ 16
#define HD_ 64
#define U_ 256
#define SCALE_ 0.125f

typedef __attribute__((ext_vector_type(16))) __bf16 v16bf;
typedef __attribute__((ext_vector_type(8)))  __bf16 v8bf;
typedef __attribute__((ext_vector_type(8)))  float  v8f;
typedef __attribute__((ext_vector_type(4))) unsigned int u32x4;
typedef __attribute__((ext_vector_type(8))) int i32x8;
typedef __attribute__((ext_vector_type(4))) int i32x4;

union Frag { v16bf v; v8bf h[2]; };

__device__ __forceinline__ v8f vzero8() {
  v8f z = {0.f,0.f,0.f,0.f,0.f,0.f,0.f,0.f};
  return z;
}
__device__ __forceinline__ v8f wmma_bf16(const Frag a, const Frag b, v8f c) {
  // D = A(16x32 bf16) * B(32x16 bf16) + C(16x16 f32)
  return __builtin_amdgcn_wmma_f32_16x16x32_bf16(false, a.v, false, b.v,
                                                 (short)0, c, false, false);
}
__device__ __forceinline__ __bf16 f2bf(float f){ return (__bf16)f; }

// ---------------------------------------------------------------------------
// TDM: DMA a 128-row x 32-element bf16 tile (row stride K elements) from
// global memory into LDS at byte offset lds_off. D# per CDNA5 ISA ch.8:
//   group0: [1:0]=count=1 | [63:32]=lds_addr | [120:64]=global_addr | [127:126]=type=2
//   group1: [17:16]=data_size(1->2B) | [79:48]=tensor_dim0=K | [111:80]=tensor_dim1=128
//           [127:112]=tile_dim0=32 | [143:128]=tile_dim1=128 | [207:160]=dim0_stride=K
//   groups 2/3: zero (2-D tensor)
__device__ __forceinline__ void tdm_load_tile_128x32(unsigned lds_off,
                                                     unsigned long long gaddr,
                                                     int K){
  u32x4 g0;
  g0[0] = 1u;                                            // count=1, user mode
  g0[1] = lds_off;                                       // lds_addr (bytes)
  g0[2] = (unsigned)(gaddr & 0xffffffffu);               // global_addr[31:0]
  g0[3] = (unsigned)((gaddr >> 32) & 0x01ffffffu) | 0x80000000u; // [56:32] | type=2
  i32x8 g1;
  g1[0] = 0x00010000;                                    // data_size = 2 bytes
  g1[1] = (int)(((unsigned)K & 0xffffu) << 16);          // tensor_dim0[15:0]
  g1[2] = (int)((((unsigned)K >> 16) & 0xffffu) | (128u << 16)); // dim0 hi | dim1 lo
  g1[3] = (int)(32u << 16);                              // dim1 hi=0 | tile_dim0=32
  g1[4] = 128;                                           // tile_dim1=128 | tile_dim2=0
  g1[5] = K;                                             // tensor_dim0_stride[31:0]
  g1[6] = 0;                                             // stride0 hi | stride1 lo
  g1[7] = 0;
  i32x4 z4 = {0,0,0,0};
#if __clang_major__ >= 23
  i32x8 z8 = {0,0,0,0,0,0,0,0};
  __builtin_amdgcn_tensor_load_to_lds(g0, g1, z4, z4, z8, 0);
#else
  __builtin_amdgcn_tensor_load_to_lds(g0, g1, z4, z4, 0);
#endif
}

// ---------------------------------------------------------------------------
// Elementwise f32 -> bf16
__global__ void cvt_bf16_kernel(const float* __restrict__ in, __bf16* __restrict__ out, int n){
  int i = blockIdx.x*blockDim.x + threadIdx.x;
  if (i < n) out[i] = f2bf(in[i]);
}

// Transpose [K][N] f32 -> [N][K] bf16 (weights pre-transposed so WMMA B-fragments
// are contiguous per-lane loads). Dims multiples of 32.
__global__ void transpose_cvt_kernel(const float* __restrict__ in, __bf16* __restrict__ out,
                                     int K, int N){
  __shared__ float t[32][33];
  const int k0 = blockIdx.y*32, n0 = blockIdx.x*32;
  for (int i = threadIdx.y; i < 32; i += 8)
    t[i][threadIdx.x] = in[(size_t)(k0+i)*N + n0 + threadIdx.x];
  __syncthreads();
  for (int i = threadIdx.y; i < 32; i += 8)
    out[(size_t)(n0+i)*K + k0 + threadIdx.x] = f2bf(t[threadIdx.x][i]);
}

// rel table: rel[d+1023] = sign(d)*log1p(|d|)
__global__ void rel_kernel(float* __restrict__ rel){
  int i = blockIdx.x*blockDim.x + threadIdx.x;
  if (i < 2*S_-1){
    int d = i - (S_-1);
    float sg = (d>0) ? 1.f : ((d<0) ? -1.f : 0.f);
    rel[i] = sg * log1pf(fabsf((float)d));
  }
}

// mean over S: mx[b][d]
__global__ void mean_x_kernel(const float* __restrict__ x, float* __restrict__ mx){
  int i = blockIdx.x*blockDim.x + threadIdx.x;      // 0..B*D-1
  int b = i >> 10, d = i & (D_-1);
  float s = 0.f;
  const float* p = x + (size_t)b*S_*D_ + d;
  for (int ss = 0; ss < S_; ++ss) s += p[(size_t)ss*D_];
  mx[i] = s * (1.f/S_);
}

// uv[b][n] = user_emb[b] . Wuv[:,n] + buv[n]   (user_out == uv broadcast, exactly)
__global__ void uv_kernel(const float* __restrict__ ue, const float* __restrict__ W,
                          const float* __restrict__ bi, float* __restrict__ uv){
  int i = blockIdx.x*blockDim.x + threadIdx.x;      // 0..B*D-1
  int b = i >> 10, n = i & (D_-1);
  float s = bi[n];
  const float* u = ue + b*U_;
  for (int k = 0; k < U_; ++k) s += u[k]*W[(size_t)k*D_ + n];
  uv[i] = s;
}

// g_pre[b][n] = [mean_x ; user_emb] @ Wg1 + bg1
__global__ void gpre_kernel(const float* __restrict__ mx, const float* __restrict__ ue,
                            const float* __restrict__ Wg1, const float* __restrict__ bg1,
                            float* __restrict__ gp){
  int i = blockIdx.x*blockDim.x + threadIdx.x;      // 0..B*D-1
  int b = i >> 10, n = i & (D_-1);
  float s = bg1[n];
  const float* m = mx + b*D_;
  for (int k = 0; k < D_; ++k) s += m[k]*Wg1[(size_t)k*D_ + n];
  const float* u = ue + b*U_;
  for (int k = 0; k < U_; ++k) s += u[k]*Wg1[(size_t)(D_+k)*D_ + n];
  gp[i] = s;
}

// gate[b][h] = sigmoid(g[b] . Wg2[:,h] + bg2[h])
__global__ void gate_kernel(const float* __restrict__ g, const float* __restrict__ Wg2,
                            const float* __restrict__ bg2, float* __restrict__ gate){
  int i = threadIdx.x;                              // 0..63
  int b = i >> 4, hh = i & (H_-1);
  float s = bg2[hh];
  const float* gr = g + b*D_;
  for (int k = 0; k < D_; ++k) s += gr[k]*Wg2[(size_t)k*H_ + hh];
  gate[i] = 1.f/(1.f + __expf(-s));
}

// c[token][D + j] = bf16(uv[b][j])   (user attention collapses to uv broadcast)
__global__ void fill_user_kernel(const float* __restrict__ uv, __bf16* __restrict__ C){
  int i = blockIdx.x*blockDim.x + threadIdx.x;      // 0..B*S*D-1
  int tok = i >> 10, j = i & (D_-1);
  int b = tok >> 10;
  C[(size_t)tok*(2*D_) + D_ + j] = f2bf(uv[b*D_ + j]);
}

// LayerNorm(+ReLU) per row of n columns
template<typename OutT>
__global__ void ln_relu_kernel(const float* __restrict__ in, const float* __restrict__ ga,
                               const float* __restrict__ be, OutT* __restrict__ out, int n){
  __shared__ float r1[256], r2[256];
  __shared__ float sm, sr;
  const int row = blockIdx.x;
  const float* rp = in + (size_t)row*n;
  float s = 0.f, s2 = 0.f;
  for (int j = threadIdx.x; j < n; j += 256){ float v = rp[j]; s += v; s2 += v*v; }
  r1[threadIdx.x] = s; r2[threadIdx.x] = s2;
  __syncthreads();
  for (int o = 128; o > 0; o >>= 1){
    if (threadIdx.x < o){ r1[threadIdx.x] += r1[threadIdx.x+o]; r2[threadIdx.x] += r2[threadIdx.x+o]; }
    __syncthreads();
  }
  if (threadIdx.x == 0){
    float m = r1[0]/n;
    float var = r2[0]/n - m*m;
    sm = m; sr = rsqrtf(var + 1e-5f);
  }
  __syncthreads();
  OutT* op = out + (size_t)row*n;
  for (int j = threadIdx.x; j < n; j += 256){
    float v = (rp[j]-sm)*sr*ga[j] + be[j];
    v = fmaxf(v, 0.f);
    op[j] = (OutT)v;
  }
}

// ---------------------------------------------------------------------------
// Register-blocked WMMA GEMM with TDM-staged, double-buffered B tiles.
//   A:  [M][K] bf16 row-major (lda == K), fragments straight from global (L2).
//   Bt: [N][K] bf16 (weights transposed). The block's shared 128x32 B tile is
//       DMA'd into LDS by the Tensor Data Mover (wave 0 issues; TENSORcnt
//       tracks completion), double-buffered so the DMA overlaps WMMA work.
//   out mode 0: bf16 [M][ldo]; mode 1: f32 [M][ldo];
//   mode 2: bf16 v-transposed layout vt[(b*D + col)*S + s] (row = b*S+s)
// Block = 128 threads = 4 waves (2x2), each wave owns a 64x64 f32 accumulator.
__global__ __launch_bounds__(128)
void gemm_wmma_kernel(const __bf16* __restrict__ A, const __bf16* __restrict__ Bt,
                      const float* __restrict__ bias, void* __restrict__ out,
                      int K, int ldo, int mode)
{
  __shared__ __align__(16) __bf16 bstage[2][128*32];   // 16 KB double buffer
  const int lane  = threadIdx.x & 31;
  const int wave  = threadIdx.x >> 5;
  const int lhalf = lane >> 4, lmod = lane & 15;
  const int m0   = blockIdx.y*128 + (wave >> 1)*64;
  const int nblk = blockIdx.x*128;
  const int n0w  = (wave & 1)*64;

  const int nk = K >> 5;
  const unsigned long long gbase =
      (unsigned long long)(uintptr_t)(Bt + (size_t)nblk*K);
  const unsigned lds0 = (unsigned)(uintptr_t)(&bstage[0][0]);
  const unsigned lds1 = (unsigned)(uintptr_t)(&bstage[1][0]);

  if (wave == 0)                                        // tensor ops ignore EXEC;
    tdm_load_tile_128x32(lds0, gbase, K);               // issue once (uniform branch)

  v8f acc[4][4];
  for (int i = 0; i < 4; ++i) for (int j = 0; j < 4; ++j) acc[i][j] = vzero8();

  for (int it = 0; it < nk; ++it){
    if (wave == 0){
      if (it + 1 < nk){
        // issue next tile, then allow exactly 1 outstanding: tile `it` is done
        tdm_load_tile_128x32(((it+1)&1) ? lds1 : lds0,
                             gbase + (unsigned long long)(it+1)*64ull, K);
        __builtin_amdgcn_s_wait_tensorcnt((short)1);
      } else {
        __builtin_amdgcn_s_wait_tensorcnt((short)0);
      }
    }
    __syncthreads();                                    // publish tile `it`

    const int kk = it << 5;
    Frag a[4], b[4];
    for (int mt = 0; mt < 4; ++mt){
      const __bf16* p = A + (size_t)(m0 + mt*16 + lmod)*K + kk + lhalf*8;
      a[mt].h[0] = *(const v8bf*)p;
      a[mt].h[1] = *(const v8bf*)(p + 16);
    }
    const __bf16* bs = bstage[it & 1];
    for (int nt = 0; nt < 4; ++nt){
      const __bf16* p = bs + (size_t)(n0w + nt*16 + lmod)*32 + lhalf*8;
      b[nt].h[0] = *(const v8bf*)p;
      b[nt].h[1] = *(const v8bf*)(p + 16);
    }
    if (kk + 32 < K)
      __builtin_prefetch(A + (size_t)(m0 + lmod)*K + kk + 32, 0, 1);

    for (int mt = 0; mt < 4; ++mt)
      for (int nt = 0; nt < 4; ++nt)
        acc[mt][nt] = wmma_bf16(a[mt], b[nt], acc[mt][nt]);

    __syncthreads();               // all reads of buf[it&1] done before reuse
  }

  for (int mt = 0; mt < 4; ++mt)
    for (int nt = 0; nt < 4; ++nt){
      const int col = nblk + n0w + nt*16 + lmod;
      const float bv = bias ? bias[col] : 0.f;
      for (int v = 0; v < 8; ++v){
        const int row = m0 + mt*16 + lhalf*8 + v;
        const float val = acc[mt][nt][v] + bv;
        if (mode == 0)
          ((__bf16*)out)[(size_t)row*ldo + col] = f2bf(val);
        else if (mode == 1)
          ((float*)out)[(size_t)row*ldo + col] = val;
        else {
          const int bb = row >> 10, s = row & (S_-1);
          ((__bf16*)out)[((size_t)bb*D_ + col)*S_ + s] = f2bf(val);
        }
      }
    }
}

// ---------------------------------------------------------------------------
// Flash-style item attention. 2 waves per block; each wave = one 64-row q-tile
// of one (b,h). Q fragments hoisted; scores go through LDS for the C-fragment
// -> A-fragment layout change; streaming softmax with running max/sum.
__global__ __launch_bounds__(64)
void attn_kernel(const __bf16* __restrict__ Q, const __bf16* __restrict__ Kb,
                 const __bf16* __restrict__ Vt, const float* __restrict__ gate,
                 const float* __restrict__ rel, const unsigned char* __restrict__ mask,
                 __bf16* __restrict__ C)
{
  __shared__ __align__(16) float sc[2][64*64];   // scores f32, reused in-place for P bf16
  __shared__ float stM[2][64], stL[2][64], stA[2][64];

  const int lane  = threadIdx.x & 31;
  const int wave  = threadIdx.x >> 5;
  const int lhalf = lane >> 4, lmod = lane & 15;
  const int tile  = blockIdx.x*2 + wave;          // 0..1023
  const int qt = tile & 15;
  const int bh = tile >> 4;
  const int h  = bh & (H_-1);
  const int b  = bh >> 4;
  const int q0 = qt*64;

  float* scW = sc[wave];
  float* Mw = stM[wave]; float* Lw = stL[wave]; float* Aw = stA[wave];

  for (int r = lane; r < 64; r += 32){ Mw[r] = -1e30f; Lw[r] = 0.f; }

  // Hoisted Q fragments: 4 m-tiles x 2 k-steps (HD=64)
  Frag qa[4][2];
  for (int mt = 0; mt < 4; ++mt){
    const size_t tok = (size_t)(b*S_ + q0 + mt*16 + lmod);
    const __bf16* base = Q + tok*D_ + h*HD_;
    for (int ks = 0; ks < 2; ++ks){
      const __bf16* p = base + ks*32 + lhalf*8;
      qa[mt][ks].h[0] = *(const v8bf*)p;
      qa[mt][ks].h[1] = *(const v8bf*)(p + 16);
    }
  }
  v8f O[4][4];
  for (int i = 0; i < 4; ++i) for (int j = 0; j < 4; ++j) O[i][j] = vzero8();
  const float gt = gate[b*H_ + h];
  __syncthreads();

  for (int kt = 0; kt < 16; ++kt){
    const int k0 = kt*64;
    // ---- scores = Q @ K^T  (K fragments: key rows are contiguous features)
    for (int nt = 0; nt < 4; ++nt){
      Frag kb[2];
      const size_t ktok = (size_t)(b*S_ + k0 + nt*16 + lmod);
      const __bf16* base = Kb + ktok*D_ + h*HD_ + lhalf*8;
      kb[0].h[0] = *(const v8bf*)(base);
      kb[0].h[1] = *(const v8bf*)(base + 16);
      kb[1].h[0] = *(const v8bf*)(base + 32);
      kb[1].h[1] = *(const v8bf*)(base + 48);
      for (int mt = 0; mt < 4; ++mt){
        v8f acc = vzero8();
        acc = wmma_bf16(qa[mt][0], kb[0], acc);
        acc = wmma_bf16(qa[mt][1], kb[1], acc);
        float* dst = scW + (size_t)(mt*16 + lhalf*8)*64 + nt*16 + lmod;
        for (int v = 0; v < 8; ++v) dst[(size_t)v*64] = acc[v];
      }
    }
    __syncthreads();
    // ---- softmax row pass (each lane owns 2 rows); P written in place as bf16
    for (int r = lane; r < 64; r += 32){
      float* row = scW + (size_t)r*64;
      const int qg = q0 + r;
      float mx = -1e30f;
      for (int j = 0; j < 64; ++j){
        const int kg = k0 + j;
        float sv = row[j]*SCALE_ + gt*rel[kg - qg + (S_-1)];
        if (!mask[b*S_ + kg]) sv = -1.0e9f;
        row[j] = sv;
        mx = fmaxf(mx, sv);
      }
      const float mold = Mw[r];
      const float mnew = fmaxf(mold, mx);
      const float alpha = __expf(mold - mnew);
      float ls = 0.f;
      __bf16* prow = (__bf16*)row;                // in-place: write 2B behind 4B read
      for (int j = 0; j < 64; ++j){
        float pv = __expf(row[j] - mnew);
        ls += pv;
        prow[j] = f2bf(pv);
      }
      Mw[r] = mnew;
      Lw[r] = Lw[r]*alpha + ls;
      Aw[r] = alpha;
    }
    __syncthreads();
    // ---- rescale running output
    for (int mt = 0; mt < 4; ++mt)
      for (int v = 0; v < 8; ++v){
        const float al = Aw[mt*16 + lhalf*8 + v];
        for (int dt = 0; dt < 4; ++dt) O[mt][dt][v] *= al;
      }
    // ---- O += P @ V   (V in vt[d][s] layout -> contiguous B-fragments)
    const __bf16* pb = (const __bf16*)scW;        // P row stride = 128 bf16
    for (int ks = 0; ks < 2; ++ks){
      Frag pf[4];
      for (int mt = 0; mt < 4; ++mt){
        const __bf16* p = pb + (size_t)(mt*16 + lmod)*128 + ks*32 + lhalf*8;
        pf[mt].h[0] = *(const v8bf*)p;
        pf[mt].h[1] = *(const v8bf*)(p + 16);
      }
      for (int dt = 0; dt < 4; ++dt){
        Frag vb;
        const size_t drow = (size_t)(b*D_ + h*HD_ + dt*16 + lmod);
        const __bf16* p = Vt + drow*S_ + k0 + ks*32 + lhalf*8;
        vb.h[0] = *(const v8bf*)p;
        vb.h[1] = *(const v8bf*)(p + 16);
        for (int mt = 0; mt < 4; ++mt)
          O[mt][dt] = wmma_bf16(pf[mt], vb, O[mt][dt]);
      }
    }
    __syncthreads();
  }
  // ---- normalize and write item half of c
  for (int mt = 0; mt < 4; ++mt)
    for (int v = 0; v < 8; ++v){
      const int r = mt*16 + lhalf*8 + v;
      const float il = 1.f / Lw[r];
      const size_t tok = (size_t)(b*S_ + q0 + r);
      for (int dt = 0; dt < 4; ++dt){
        const int col = h*HD_ + dt*16 + lmod;
        C[tok*(2*D_) + col] = f2bf(O[mt][dt][v] * il);
      }
    }
}

// ---------------------------------------------------------------------------
extern "C" void kernel_launch(void* const* d_in, const int* in_sizes, int n_in,
                              void* d_out, int out_size, void* d_ws, size_t ws_size,
                              hipStream_t stream)
{
  (void)in_sizes; (void)n_in; (void)out_size; (void)ws_size;

  const float* x    = (const float*)d_in[0];
  const float* ue   = (const float*)d_in[1];
  const unsigned char* mask = (const unsigned char*)d_in[2];
  // d_in[3..6] (Wuq,buq,Wuk,buk) unused: user attention collapses algebraically.
  const float* Wuv  = (const float*)d_in[7];
  const float* buv  = (const float*)d_in[8];
  const float* Wiq  = (const float*)d_in[9];
  const float* biq  = (const float*)d_in[10];
  const float* Wik  = (const float*)d_in[11];
  const float* bik  = (const float*)d_in[12];
  const float* Wiv  = (const float*)d_in[13];
  const float* biv  = (const float*)d_in[14];
  const float* Wg1  = (const float*)d_in[15];
  const float* bg1  = (const float*)d_in[16];
  const float* glng = (const float*)d_in[17];
  const float* glnb = (const float*)d_in[18];
  const float* Wg2  = (const float*)d_in[19];
  const float* bg2  = (const float*)d_in[20];
  const float* Wo1  = (const float*)d_in[21];
  const float* bo1  = (const float*)d_in[22];
  const float* olng = (const float*)d_in[23];
  const float* olnb = (const float*)d_in[24];
  const float* Wo2  = (const float*)d_in[25];
  const float* bo2  = (const float*)d_in[26];
  float* out = (float*)d_out;

  // workspace bump allocator
  char* wp = (char*)d_ws;
  auto alloc = [&](size_t bytes)->char* {
    char* r = wp; wp += (bytes + 255) & ~(size_t)255; return r;
  };
  const size_t NTOK = (size_t)B_*S_;
  __bf16* x_bf   = (__bf16*)alloc(NTOK*D_*2);
  __bf16* q_bf   = (__bf16*)alloc(NTOK*D_*2);
  __bf16* k_bf   = (__bf16*)alloc(NTOK*D_*2);
  __bf16* vt_bf  = (__bf16*)alloc(NTOK*D_*2);        // vt[(b*D+col)][s]
  __bf16* wiq_t  = (__bf16*)alloc((size_t)D_*D_*2);
  __bf16* wik_t  = (__bf16*)alloc((size_t)D_*D_*2);
  __bf16* wiv_t  = (__bf16*)alloc((size_t)D_*D_*2);
  __bf16* wo1_t  = (__bf16*)alloc((size_t)2*D_*2*D_*2);
  __bf16* wo2_t  = (__bf16*)alloc((size_t)D_*2*D_*2);
  __bf16* c_bf   = (__bf16*)alloc(NTOK*2*D_*2);
  float*  hpre   = (float*)alloc(NTOK*2*D_*4);
  __bf16* c2_bf  = (__bf16*)alloc(NTOK*2*D_*2);
  float*  meanx  = (float*)alloc((size_t)B_*D_*4);
  float*  uvv    = (float*)alloc((size_t)B_*D_*4);
  float*  gpre   = (float*)alloc((size_t)B_*D_*4);
  float*  gvec   = (float*)alloc((size_t)B_*D_*4);
  float*  gatev  = (float*)alloc((size_t)B_*H_*4);
  float*  relt   = (float*)alloc((size_t)(2*S_-1)*4);

  // 1) conversions / transposes
  {
    int n = (int)(NTOK*D_);
    cvt_bf16_kernel<<<dim3(n/256), dim3(256), 0, stream>>>(x, x_bf, n);
  }
  transpose_cvt_kernel<<<dim3(D_/32, D_/32),   dim3(32,8), 0, stream>>>(Wiq, wiq_t, D_, D_);
  transpose_cvt_kernel<<<dim3(D_/32, D_/32),   dim3(32,8), 0, stream>>>(Wik, wik_t, D_, D_);
  transpose_cvt_kernel<<<dim3(D_/32, D_/32),   dim3(32,8), 0, stream>>>(Wiv, wiv_t, D_, D_);
  transpose_cvt_kernel<<<dim3(2*D_/32, 2*D_/32), dim3(32,8), 0, stream>>>(Wo1, wo1_t, 2*D_, 2*D_);
  transpose_cvt_kernel<<<dim3(D_/32, 2*D_/32), dim3(32,8), 0, stream>>>(Wo2, wo2_t, 2*D_, D_);
  rel_kernel<<<dim3(8), dim3(256), 0, stream>>>(relt);

  // 2) gate path + uv (tiny)
  mean_x_kernel<<<dim3(16), dim3(256), 0, stream>>>(x, meanx);
  uv_kernel<<<dim3(16), dim3(256), 0, stream>>>(ue, Wuv, buv, uvv);
  gpre_kernel<<<dim3(16), dim3(256), 0, stream>>>(meanx, ue, Wg1, bg1, gpre);
  ln_relu_kernel<float><<<dim3(B_), dim3(256), 0, stream>>>(gpre, glng, glnb, gvec, D_);
  gate_kernel<<<dim3(1), dim3(64), 0, stream>>>(gvec, Wg2, bg2, gatev);

  // 3) QKV projections (WMMA GEMMs); V written transposed for the PV stage
  dim3 gblk(128);
  dim3 ggrd(D_/128, (int)(NTOK/128));
  gemm_wmma_kernel<<<ggrd, gblk, 0, stream>>>(x_bf, wiq_t, biq, q_bf, D_, D_, 0);
  gemm_wmma_kernel<<<ggrd, gblk, 0, stream>>>(x_bf, wik_t, bik, k_bf, D_, D_, 0);
  gemm_wmma_kernel<<<ggrd, gblk, 0, stream>>>(x_bf, wiv_t, biv, vt_bf, D_, 0, 2);

  // 4) c = [item_attention ; uv broadcast]
  fill_user_kernel<<<dim3((int)(NTOK*D_/256)), dim3(256), 0, stream>>>(uvv, c_bf);
  attn_kernel<<<dim3(B_*H_*(S_/64)/2), dim3(64), 0, stream>>>(
      q_bf, k_bf, vt_bf, gatev, relt, mask, c_bf);

  // 5) output MLP
  gemm_wmma_kernel<<<dim3(2*D_/128, (int)(NTOK/128)), gblk, 0, stream>>>(
      c_bf, wo1_t, bo1, hpre, 2*D_, 2*D_, 1);
  ln_relu_kernel<__bf16><<<dim3((int)NTOK), dim3(256), 0, stream>>>(
      hpre, olng, olnb, c2_bf, 2*D_);
  gemm_wmma_kernel<<<dim3(D_/128, (int)(NTOK/128)), gblk, 0, stream>>>(
      c2_bf, wo2_t, bo2, out, 2*D_, D_, 1);
}
